// CdfgReader_11424613007428
// MI455X (gfx1250) — matrix operand order repaired and checked by reading, not compiled, not run on hardware
//
#include <hip/hip_runtime.h>

typedef __bf16 bf16_t;
typedef __attribute__((ext_vector_type(16))) __bf16 v16bf;
typedef __attribute__((ext_vector_type(8)))  __bf16 v8bf;
typedef __attribute__((ext_vector_type(8)))  float  v8f;

#define G_ 4
#define N_ 1024
#define F_ 256
#define H_ 256
#define B_ 32
#define L_ 4

// ---------------- WMMA fragment helpers (bf16 16x16x32, wave32) ----------------
// A-matrix 16x32 (MxK), row-major source with leading dim lda:
// lanes 0-15: row M=lane, K = [0..7] in v0-3 and [16..23] in v4-7
// lanes 16-31: row M=lane-16, K = [8..15] and [24..31]
__device__ __forceinline__ v16bf frag_a(const bf16_t* __restrict__ p0, int lda) {
  const int lane = threadIdx.x & 31;
  const int r = lane & 15, hf = lane >> 4;
  const bf16_t* p = p0 + (size_t)r * lda + hf * 8;
  v8bf lo = *(const v8bf*)(p);
  v8bf hi = *(const v8bf*)(p + 16);
  v16bf a;
#pragma unroll
  for (int i = 0; i < 8; ++i) { a[i] = lo[i]; a[i + 8] = hi[i]; }
  return a;
}

// B-matrix 32x16 (KxN), source stored N-major ([n][k], leading dim ldb):
// lanes 0-15: col N=lane, K = [0..7] and [16..23]; lanes 16-31: K = [8..15] and [24..31]
__device__ __forceinline__ v16bf frag_b(const bf16_t* __restrict__ p0, int ldb) {
  const int lane = threadIdx.x & 31;
  const int r = lane & 15, hf = lane >> 4;
  const bf16_t* p = p0 + (size_t)r * ldb + hf * 8;
  v8bf lo = *(const v8bf*)(p);
  v8bf hi = *(const v8bf*)(p + 16);
  v16bf b;
#pragma unroll
  for (int i = 0; i < 8; ++i) { b[i] = lo[i]; b[i + 8] = hi[i]; }
  return b;
}

#define WMMA_BF16(a, b, c) \
  __builtin_amdgcn_wmma_f32_16x16x32_bf16(false, (a), false, (b), (short)0, (c), false, false)

// ---------------- one-time converts ----------------
__global__ void k_cast(const float* __restrict__ src, bf16_t* __restrict__ dst, int n) {
  int i = blockIdx.x * blockDim.x + threadIdx.x;
  if (i < n) dst[i] = (bf16_t)src[i];
}

// dst[c][r] = (bf16)src[r][c]   (dst is [cols][rows])
__global__ void k_tcast(const float* __restrict__ src, bf16_t* __restrict__ dst,
                        int rows, int cols) {
  int i = blockIdx.x * blockDim.x + threadIdx.x;
  if (i < rows * cols) {
    int r = i / cols, c = i % cols;
    dst[(size_t)c * rows + r] = (bf16_t)src[(size_t)r * cols + c];
  }
}

// ---------------- input dense layer: XT/Res = relu(X @ w_in + b_in), stored [h][node] ----------------
// grid (N/64, B), block 128 (4 waves). wave w covers n (=h) range [w*64, w*64+64),
// block covers m (=node) range [mb0, mb0+64). 4x4 tile register blocking per wave.
__global__ void __launch_bounds__(128, 1)
k_input(const bf16_t* __restrict__ Xbf, const bf16_t* __restrict__ WinT,
        const float* __restrict__ b_in, const int* __restrict__ gidx,
        bf16_t* __restrict__ XT, float* __restrict__ ResT) {
  const int b = blockIdx.y;
  const int g = gidx[b];
  const int mb0 = blockIdx.x * 64;
  const int w = threadIdx.x >> 5;
  const int lane = threadIdx.x & 31;
  const int r = lane & 15, hf = lane >> 4;
  const bf16_t* Abase = Xbf + (size_t)g * N_ * F_;
  v8f acc[4][4] = {};
  for (int kb = 0; kb < F_; kb += 32) {
    v16bf af[4];
#pragma unroll
    for (int mi = 0; mi < 4; ++mi)
      af[mi] = frag_a(Abase + (size_t)(mb0 + mi * 16) * F_ + kb, F_);
#pragma unroll
    for (int t = 0; t < 4; ++t) {
      const int nb = w * 64 + t * 16;
      v16bf bfv = frag_b(WinT + (size_t)nb * F_ + kb, F_);
#pragma unroll
      for (int mi = 0; mi < 4; ++mi)
        acc[mi][t] = WMMA_BF16(af[mi], bfv, acc[mi][t]);
    }
  }
#pragma unroll
  for (int t = 0; t < 4; ++t) {
    const int h = w * 64 + t * 16 + r;
    const float bias = b_in[h];
#pragma unroll
    for (int mi = 0; mi < 4; ++mi) {
      const size_t off = ((size_t)b * H_ + h) * N_ + mb0 + mi * 16 + hf * 8;
      v8bf ob; v8f of;
#pragma unroll
      for (int i = 0; i < 8; ++i) {
        float v = fmaxf(acc[mi][t][i] + bias, 0.0f);
        of[i] = v; ob[i] = (bf16_t)v;
      }
      *(v8bf*)(XT + off) = ob;
      *(v8f*)(ResT + off) = of;
    }
  }
}

// ---------------- fused GCN layer ----------------
// Phase 1: t_slab[64][256] = Adj[64 rows] @ x   (A-op row-major adjacency, B-op XTin [h][node])
//          -> scattered into LDS row-major [node][h]
// Phase 2: y = act(t_slab @ W + b)  (A-op = LDS slab row-major, B-op = WT [h_out][h_in])
//          non-final: relu -> XTout (natural [h][node] bf16). final: tanh + residual -> ResT.
// XT is ping-ponged across layers because phase 1 reads all of XTin while phase 2 writes XTout.
__global__ void __launch_bounds__(128, 1)
k_layer(const bf16_t* __restrict__ Abf, const bf16_t* __restrict__ XTin,
        const int* __restrict__ gidx, const bf16_t* __restrict__ WTl,
        const float* __restrict__ bias_l, int is_final,
        bf16_t* __restrict__ XTout, float* __restrict__ ResT) {
  __shared__ bf16_t tls[64 * H_];  // 32 KB slab: t rows [mb0, mb0+64), all 256 h
  const int b = blockIdx.y;
  const int g = gidx[b];
  const int mb0 = blockIdx.x * 64;
  const int w = threadIdx.x >> 5;
  const int lane = threadIdx.x & 31;
  const int r = lane & 15, hf = lane >> 4;

  // -------- phase 1: Adj @ x --------
  {
    const bf16_t* Abase = Abf + (size_t)g * N_ * N_;
    const bf16_t* Bbase = XTin + (size_t)b * H_ * N_;
    v8f acc[4][4] = {};
    for (int kb = 0; kb < N_; kb += 32) {
      v16bf af[4];
#pragma unroll
      for (int mi = 0; mi < 4; ++mi)
        af[mi] = frag_a(Abase + (size_t)(mb0 + mi * 16) * N_ + kb, N_);
#pragma unroll
      for (int t = 0; t < 4; ++t) {
        const int nb = w * 64 + t * 16;
        v16bf bfv = frag_b(Bbase + (size_t)nb * N_ + kb, N_);
#pragma unroll
        for (int mi = 0; mi < 4; ++mi)
          acc[mi][t] = WMMA_BF16(af[mi], bfv, acc[mi][t]);
      }
    }
    // scatter D (n-major fragments) into LDS row-major [node][h]
#pragma unroll
    for (int t = 0; t < 4; ++t) {
      const int h = w * 64 + t * 16 + r;
#pragma unroll
      for (int mi = 0; mi < 4; ++mi) {
#pragma unroll
        for (int i = 0; i < 8; ++i) {
          const int lm = mi * 16 + hf * 8 + i;  // local node row 0..63
          tls[lm * H_ + h] = (bf16_t)acc[mi][t][i];
        }
      }
    }
  }
  __syncthreads();

  // -------- phase 2: t_slab @ W --------
  {
    v8f acc2[4][4] = {};
    for (int kb = 0; kb < H_; kb += 32) {
      v16bf af[4];
#pragma unroll
      for (int mi = 0; mi < 4; ++mi)
        af[mi] = frag_a(tls + (size_t)(mi * 16) * H_ + kb, H_);  // contiguous LDS b128 pairs
#pragma unroll
      for (int t = 0; t < 4; ++t) {
        const int nb = w * 64 + t * 16;
        v16bf bfv = frag_b(WTl + (size_t)nb * H_ + kb, H_);
#pragma unroll
        for (int mi = 0; mi < 4; ++mi)
          acc2[mi][t] = WMMA_BF16(af[mi], bfv, acc2[mi][t]);
      }
    }
#pragma unroll
    for (int t = 0; t < 4; ++t) {
      const int h = w * 64 + t * 16 + r;
      const float bias = bias_l[h];
#pragma unroll
      for (int mi = 0; mi < 4; ++mi) {
        const size_t off = ((size_t)b * H_ + h) * N_ + mb0 + mi * 16 + hf * 8;
        if (!is_final) {
          v8bf ob;
#pragma unroll
          for (int i = 0; i < 8; ++i) ob[i] = (bf16_t)fmaxf(acc2[mi][t][i] + bias, 0.0f);
          *(v8bf*)(XTout + off) = ob;
        } else {
          v8f res = *(const v8f*)(ResT + off);
          v8f of;
#pragma unroll
          for (int i = 0; i < 8; ++i) of[i] = tanhf(acc2[mi][t][i] + bias) + res[i];
          *(v8f*)(ResT + off) = of;
        }
      }
    }
  }
}

// ---------------- masked mean over nodes: one wave per (b, h) row of ResT[b][h][:] ----------------
__global__ void k_reduce(const float* __restrict__ ResT, const unsigned char* __restrict__ mask,
                         float* __restrict__ out) {
  const int wid = (blockIdx.x * blockDim.x + threadIdx.x) >> 5;
  if (wid >= B_ * H_) return;
  const int b = wid / H_, h = wid % H_;
  const int lane = threadIdx.x & 31;
  const float* row = ResT + ((size_t)b * H_ + h) * N_;
  const unsigned char* mrow = mask + (size_t)b * N_;
  float s = 0.0f, c = 0.0f;
  for (int n = lane; n < N_; n += 32) {
    const float mv = mrow[n] ? 1.0f : 0.0f;
    s += row[n] * mv;
    c += mv;
  }
#pragma unroll
  for (int o = 16; o > 0; o >>= 1) {
    s += __shfl_down(s, o, 32);
    c += __shfl_down(c, o, 32);
  }
  if (lane == 0) out[(size_t)b * H_ + h] = s / fmaxf(c, 1.0f);
}

extern "C" void kernel_launch(void* const* d_in, const int* in_sizes, int n_in,
                              void* d_out, int out_size, void* d_ws, size_t ws_size,
                              hipStream_t stream) {
  (void)in_sizes; (void)n_in; (void)out_size; (void)ws_size;
  const float* batch_xs = (const float*)d_in[0];
  const float* batch_as = (const float*)d_in[1];
  const float* w_in     = (const float*)d_in[2];
  const float* b_in     = (const float*)d_in[3];
  const float* gcn_w    = (const float*)d_in[4];
  const float* gcn_b    = (const float*)d_in[5];
  const int*   gidx     = (const int*)d_in[6];
  const unsigned char* mask = (const unsigned char*)d_in[7];
  float* out = (float*)d_out;

  char* ws = (char*)d_ws;
  size_t off = 0;
  bf16_t* Abf  = (bf16_t*)(ws + off); off += (size_t)G_ * N_ * N_ * 2;   // 8 MB
  bf16_t* Xbf  = (bf16_t*)(ws + off); off += (size_t)G_ * N_ * F_ * 2;   // 2 MB
  bf16_t* WinT = (bf16_t*)(ws + off); off += (size_t)H_ * F_ * 2;        // 128 KB
  bf16_t* WT   = (bf16_t*)(ws + off); off += (size_t)L_ * H_ * H_ * 2;   // 512 KB
  bf16_t* XT0  = (bf16_t*)(ws + off); off += (size_t)B_ * H_ * N_ * 2;   // 16 MB
  bf16_t* XT1  = (bf16_t*)(ws + off); off += (size_t)B_ * H_ * N_ * 2;   // 16 MB
  float*  ResT = (float*)(ws + off);  off += (size_t)B_ * H_ * N_ * 4;   // 32 MB

  k_cast<<<(G_ * N_ * N_) / 256, 256, 0, stream>>>(batch_as, Abf, G_ * N_ * N_);
  k_cast<<<(G_ * N_ * F_) / 256, 256, 0, stream>>>(batch_xs, Xbf, G_ * N_ * F_);
  k_tcast<<<(F_ * H_) / 256, 256, 0, stream>>>(w_in, WinT, F_, H_);
  for (int l = 0; l < L_; ++l)
    k_tcast<<<(H_ * H_) / 256, 256, 0, stream>>>(gcn_w + (size_t)l * H_ * H_,
                                                 WT + (size_t)l * H_ * H_, H_, H_);

  k_input<<<dim3(N_ / 64, B_), 128, 0, stream>>>(Xbf, WinT, b_in, gidx, XT0, ResT);
  const bf16_t* xin = XT0;
  bf16_t* xout = XT1;
  for (int l = 0; l < L_; ++l) {
    k_layer<<<dim3(N_ / 64, B_), 128, 0, stream>>>(Abf, xin, gidx,
                                                   WT + (size_t)l * H_ * H_,
                                                   gcn_b + (size_t)l * H_,
                                                   (l == L_ - 1) ? 1 : 0, xout, ResT);
    const bf16_t* tmp = xout;
    xout = (bf16_t*)xin;
    xin = tmp;
  }
  k_reduce<<<(B_ * H_ * 32) / 256, 256, 0, stream>>>(ResT, mask, out);
}
